// EnergyFunctionCUDA_85555748536986
// MI455X (gfx1250) — matrix operand order repaired and checked by reading, not compiled, not run on hardware
//
#include <hip/hip_runtime.h>

typedef float  v2f  __attribute__((ext_vector_type(2)));
typedef float  v4f  __attribute__((ext_vector_type(4)));
typedef float  v8f  __attribute__((ext_vector_type(8)));
typedef __bf16 v4bf __attribute__((ext_vector_type(4)));
typedef __bf16 v8bf __attribute__((ext_vector_type(8)));
typedef __bf16 v16bf __attribute__((ext_vector_type(16)));

#define N_    4096
#define D_    512
#define M_    50000
#define K_    32
#define COLT  (M_ / 16)                    // 3125 column tiles of mu
#define ROWT  (N_ / 16)                    // 256 row tiles of x
#define WAVES 8
#define ITERS ((COLT + WAVES - 1) / WAVES) // 391
#define NPART ((ROWT * ROWT) / WAVES)      // 8192 geom partials

// ---------------------------------------------------------------------------
// Split fp32 -> (bf16 hi, bf16 lo) for bf16x3 fp32-emulated GEMM.
// ---------------------------------------------------------------------------
__global__ __launch_bounds__(256) void split_bf16_kernel(
    const float* __restrict__ src, __bf16* __restrict__ hi,
    __bf16* __restrict__ lo, int count4)
{
    int i = blockIdx.x * blockDim.x + threadIdx.x;
    if (i >= count4) return;
    v4f f = *(const v4f*)(src + 4 * (size_t)i);
    v4bf h, l;
    #pragma unroll
    for (int j = 0; j < 4; ++j) {
        float fj = f[j];
        __bf16 hj = (__bf16)fj;
        h[j] = hj;
        l[j] = (__bf16)(fj - (float)hj);
    }
    *(v4bf*)(hi + 4 * (size_t)i) = h;
    *(v4bf*)(lo + 4 * (size_t)i) = l;
}

// ---------------------------------------------------------------------------
// Tile helpers: one wave computes a 16x16 fp32 tile of  Arows @ Brows^T.
// ---------------------------------------------------------------------------
__device__ __forceinline__ v16bf cat16(v8bf a, v8bf b)
{
    return __builtin_shufflevector(a, b, 0, 1, 2, 3, 4, 5, 6, 7,
                                         8, 9, 10, 11, 12, 13, 14, 15);
}

// f32 path: V_WMMA_F32_16X16X4_F32; arow/brow pre-offset by +2*half.
__device__ __forceinline__ v8f tile_f32(const float* __restrict__ arow,
                                        const float* __restrict__ brow)
{
    v8f c = {0.f, 0.f, 0.f, 0.f, 0.f, 0.f, 0.f, 0.f};
    #pragma unroll 4
    for (int k = 0; k < D_; k += 4) {
        v2f a = *(const v2f*)(arow + k);
        v2f b = *(const v2f*)(brow + k);
        c = __builtin_amdgcn_wmma_f32_16x16x4_f32(
                false, a, false, b, (short)0, c, false, false);
    }
    return c;
}

// bf16x3 path: hi*hi + hi*lo + lo*hi with V_WMMA_F32_16X16X32_BF16.
// ah/al/bh/bl point to the start of this lane's row (no half offset).
__device__ __forceinline__ v8f tile_bf16x3(const __bf16* __restrict__ ahr,
                                           const __bf16* __restrict__ alr,
                                           const __bf16* __restrict__ bhr,
                                           const __bf16* __restrict__ blr,
                                           int h)
{
    v8f c = {0.f, 0.f, 0.f, 0.f, 0.f, 0.f, 0.f, 0.f};
    const int aoff0 = 8 * h, aoff1 = 16 + 8 * h;   // A: K=8h+{0..7}, 16+8h+{0..7}
    const int boff  = 8 * h;                       // B: K=8h+{0..7}, 16+8h+{0..7}
    #pragma unroll 2
    for (int kb = 0; kb < D_; kb += 32) {
        v8bf ah0 = *(const v8bf*)(ahr + kb + aoff0);
        v8bf ah1 = *(const v8bf*)(ahr + kb + aoff1);
        v8bf al0 = *(const v8bf*)(alr + kb + aoff0);
        v8bf al1 = *(const v8bf*)(alr + kb + aoff1);
        v8bf bh0 = *(const v8bf*)(bhr + kb + boff);
        v8bf bh1 = *(const v8bf*)(bhr + kb + boff + 16);
        v8bf bl0 = *(const v8bf*)(blr + kb + boff);
        v8bf bl1 = *(const v8bf*)(blr + kb + boff + 16);
        v16bf ah = cat16(ah0, ah1);
        v16bf al = cat16(al0, al1);
        v16bf bh = cat16(bh0, bh1);
        v16bf bl = cat16(bl0, bl1);
        c = __builtin_amdgcn_wmma_f32_16x16x32_bf16(
                false, ah, false, bh, (short)0, c, false, false);
        c = __builtin_amdgcn_wmma_f32_16x16x32_bf16(
                false, ah, false, bl, (short)0, c, false, false);
        c = __builtin_amdgcn_wmma_f32_16x16x32_bf16(
                false, al, false, bh, (short)0, c, false, false);
    }
    return c;
}

// ---------------------------------------------------------------------------
// Kernel A: fused sims = x @ mu.T + streaming per-row top-32 + e_splat/e_comp.
// One block = 16 rows of x; 8 waves sweep the 3125 column tiles in lockstep.
// ---------------------------------------------------------------------------
template <bool BF>
__global__ __launch_bounds__(256) void sims_topk_kernel(
    const float* __restrict__ x, const float* __restrict__ mu,
    const __bf16* __restrict__ xhi, const __bf16* __restrict__ xlo,
    const __bf16* __restrict__ muhi, const __bf16* __restrict__ mulo,
    const float* __restrict__ alpha, const float* __restrict__ kappa,
    const float* __restrict__ Ww, const float* __restrict__ Wb,
    float* __restrict__ out)
{
    __shared__ float s_sims[WAVES][16][17];   // +1 pad: conflict-free row reads
    __shared__ float s_tval[WAVES][16][K_];
    __shared__ int   s_tidx[WAVES][16][K_];
    __shared__ float s_mval[16][K_];
    __shared__ int   s_midx[16][K_];

    const int tid  = threadIdx.x;
    const int wave = tid >> 5;
    const int lane = tid & 31;
    const int half = lane >> 4;
    const int lr   = lane & 15;
    const int row_base = blockIdx.x * 16;
    const size_t arow = (size_t)(row_base + lr) * D_;

    for (int i = tid; i < WAVES * 16 * K_; i += 256) {
        ((float*)s_tval)[i] = -1e30f;
        ((int*)s_tidx)[i]   = 0;
    }
    __syncthreads();

    float curMin    = -1e30f;
    int   curMinPos = 0;

    for (int it = 0; it < ITERS; ++it) {
        const int  tile   = it * WAVES + wave;
        const bool active = (tile < COLT);         // wave-uniform branch
        if (active) {
            const size_t brow = (size_t)(tile * 16 + lr) * D_;
            v8f c;
            if constexpr (BF)
                c = tile_bf16x3(xhi + arow, xlo + arow,
                                muhi + brow, mulo + brow, half);
            else
                c = tile_f32(x + arow + 2 * half, mu + brow + 2 * half);
            #pragma unroll
            for (int v = 0; v < 8; ++v)            // C: M = v + 8*half, N = lr
                s_sims[wave][v + 8 * half][lr] = c[v];
        }
        __syncthreads();
        if (active && half == 0) {
            const int r = lr;                      // lane r maintains row r
            for (int cc = 0; cc < 16; ++cc) {
                float val = s_sims[wave][r][cc];
                if (val > curMin) {                // rare after warm-up
                    s_tval[wave][r][curMinPos] = val;
                    s_tidx[wave][r][curMinPos] = tile * 16 + cc;
                    float m = s_tval[wave][r][0]; int mp = 0;
                    for (int j = 1; j < K_; ++j) {
                        float tv = s_tval[wave][r][j];
                        if (tv < m) { m = tv; mp = j; }
                    }
                    curMin = m; curMinPos = mp;
                }
            }
        }
        __syncthreads();
    }

    // ---- merge the 8 per-wave lists and compute per-row energies ----
    if (tid < 16) {
        const int r = tid;
        float m = 1e30f; int mp = 0;
        for (int j = 0; j < K_; ++j) {
            float v0 = s_tval[0][r][j];
            s_mval[r][j] = v0; s_midx[r][j] = s_tidx[0][r][j];
            if (v0 < m) { m = v0; mp = j; }
        }
        for (int w = 1; w < WAVES; ++w)
            for (int j = 0; j < K_; ++j) {
                float v = s_tval[w][r][j];
                if (v > m) {
                    s_mval[r][mp] = v; s_midx[r][mp] = s_tidx[w][r][j];
                    m = s_mval[r][0]; mp = 0;
                    for (int q = 1; q < K_; ++q) {
                        float tv = s_mval[r][q];
                        if (tv < m) { m = tv; mp = q; }
                    }
                }
            }

        float u = -1e30f, v2 = -1e30f;             // top-2 dots
        for (int j = 0; j < K_; ++j) {
            float d = s_mval[r][j];
            if (d > u)       { v2 = u; u = d; }
            else if (d > v2) { v2 = d; }
        }

        float ksum = 0.f;
        for (int j = 0; j < K_; ++j) {
            float imp = fmaxf(kappa[s_midx[r][j]], 1e-4f);
            s_tval[0][r][j] = imp;
            ksum += imp;
        }
        float wmax = -1e30f;
        for (int j = 0; j < K_; ++j) {
            float al  = alpha[s_midx[r][j]];
            float wgt = s_tval[0][r][j] / ksum;
            float wt  = al * (s_mval[r][j] - 1.0f) * 10.0f    // /TEMP
                        + __logf(fmaxf(wgt, 1e-8f));
            s_tval[1][r][j] = wt;
            wmax = fmaxf(wmax, wt);
        }
        float se = 0.f;
        for (int j = 0; j < K_; ++j) se += __expf(s_tval[1][r][j] - wmax);
        float e_splat = -(wmax + __logf(se));

        float z = u * Ww[0] + v2 * Ww[1] + u * v2 * Ww[2] + Wb[0];
        float e_comp = 1.0f / (1.0f + __expf(-z));

        out[row_base + r] = e_splat + 0.1f * e_comp;          // LAMBDA_COMP
    }
}

// ---------------------------------------------------------------------------
// Kernel B: fused S = x @ x.T tile + masked -log(1 - min(S,1-1e-4) + 1e-4)
// reduction. One wave per 16x16 tile; deterministic per-block partials.
// ---------------------------------------------------------------------------
template <bool BF>
__global__ __launch_bounds__(256) void geom_kernel(
    const float* __restrict__ x,
    const __bf16* __restrict__ xhi, const __bf16* __restrict__ xlo,
    float* __restrict__ partials)
{
    __shared__ float s_red[WAVES];
    const int tid  = threadIdx.x;
    const int wave = tid >> 5;
    const int lane = tid & 31;
    const int half = lane >> 4;
    const int lr   = lane & 15;

    const int tileId = blockIdx.x * WAVES + wave;  // < 65536
    const int ti = tileId >> 8;
    const int tj = tileId & 255;
    const size_t arow = (size_t)(ti * 16 + lr) * D_;
    const size_t brow = (size_t)(tj * 16 + lr) * D_;

    v8f c;
    if constexpr (BF)
        c = tile_bf16x3(xhi + arow, xlo + arow, xhi + brow, xlo + brow, half);
    else
        c = tile_f32(x + arow + 2 * half, x + brow + 2 * half);

    float acc = 0.f;
    #pragma unroll
    for (int v = 0; v < 8; ++v) {
        int gm = ti * 16 + v + 8 * half;
        int gn = tj * 16 + lr;
        if (gm != gn) {
            float s = c[v];
            acc += -__logf(1.0f - fminf(s, 1.0f - 1e-4f) + 1e-4f);
        }
    }
    #pragma unroll
    for (int off = 16; off >= 1; off >>= 1)
        acc += __shfl_down(acc, off, 32);
    if (lane == 0) s_red[wave] = acc;
    __syncthreads();
    if (tid == 0) {
        float t = 0.f;
        for (int w = 0; w < WAVES; ++w) t += s_red[w];
        partials[blockIdx.x] = t;
    }
}

// ---------------------------------------------------------------------------
// Kernel C: deterministic reduction of the geom partials.
// ---------------------------------------------------------------------------
__global__ __launch_bounds__(256) void reduce_kernel(
    const float* __restrict__ partials, float* __restrict__ scalar)
{
    __shared__ float s[256];
    float a = 0.f;
    for (int i = threadIdx.x; i < NPART; i += 256) a += partials[i];
    s[threadIdx.x] = a;
    __syncthreads();
    for (int off = 128; off >= 1; off >>= 1) {
        if (threadIdx.x < off) s[threadIdx.x] += s[threadIdx.x + off];
        __syncthreads();
    }
    if (threadIdx.x == 0) scalar[0] = s[0];
}

// ---------------------------------------------------------------------------
// Kernel D: out[i] += LAMBDA_GEOM * e_geom.
// ---------------------------------------------------------------------------
__global__ void add_geom_kernel(float* __restrict__ out,
                                const float* __restrict__ scalar)
{
    int i = blockIdx.x * blockDim.x + threadIdx.x;
    if (i < N_) {
        float e_geom = scalar[0] / ((float)N_ * (float)(N_ - 1));
        out[i] += 0.1f * e_geom;
    }
}

extern "C" void kernel_launch(void* const* d_in, const int* in_sizes, int n_in,
                              void* d_out, int out_size, void* d_ws, size_t ws_size,
                              hipStream_t stream)
{
    const float* x     = (const float*)d_in[0];   // (4096, 512)
    const float* mu    = (const float*)d_in[1];   // (50000, 512)
    const float* alpha = (const float*)d_in[2];   // (50000,)
    const float* kappa = (const float*)d_in[3];   // (50000,)
    const float* Ww    = (const float*)d_in[4];   // (3,)
    const float* Wb    = (const float*)d_in[5];   // ()
    float* out = (float*)d_out;                   // (4096,)

    // Workspace layout for the bf16x3 path (hi/lo splits + geom partials).
    const size_t xsz  = (size_t)N_ * D_ * sizeof(__bf16);   // 4 MiB each
    const size_t musz = (size_t)M_ * D_ * sizeof(__bf16);   // 51.2 MB each
    const size_t psz  = (size_t)(NPART + 1) * sizeof(float);
    const size_t A256 = 255;
    const size_t o_xlo  = (xsz + A256) & ~A256;
    const size_t o_muhi = (o_xlo + xsz + A256) & ~A256;
    const size_t o_mulo = (o_muhi + musz + A256) & ~A256;
    const size_t o_part = (o_mulo + musz + A256) & ~A256;
    const size_t needed = o_part + psz;

    char* ws = (char*)d_ws;
    const bool use_bf = (ws_size >= needed);      // constant per deployment

    float* partials = use_bf ? (float*)(ws + o_part) : (float*)ws;
    float* scalar   = partials + NPART;

    if (use_bf) {
        __bf16* xhi  = (__bf16*)ws;
        __bf16* xlo  = (__bf16*)(ws + o_xlo);
        __bf16* muhi = (__bf16*)(ws + o_muhi);
        __bf16* mulo = (__bf16*)(ws + o_mulo);

        const int xc4  = (N_ * D_) / 4;           // 524288
        const int muc4 = (M_ * D_) / 4;           // 6400000
        split_bf16_kernel<<<(xc4 + 255) / 256, 256, 0, stream>>>(x, xhi, xlo, xc4);
        split_bf16_kernel<<<(muc4 + 255) / 256, 256, 0, stream>>>(mu, muhi, mulo, muc4);

        sims_topk_kernel<true><<<ROWT, 256, 0, stream>>>(
            nullptr, nullptr, xhi, xlo, muhi, mulo, alpha, kappa, Ww, Wb, out);
        geom_kernel<true><<<(ROWT * ROWT) / WAVES, 256, 0, stream>>>(
            nullptr, xhi, xlo, partials);
    } else {
        sims_topk_kernel<false><<<ROWT, 256, 0, stream>>>(
            x, mu, nullptr, nullptr, nullptr, nullptr, alpha, kappa, Ww, Wb, out);
        geom_kernel<false><<<(ROWT * ROWT) / WAVES, 256, 0, stream>>>(
            x, nullptr, nullptr, partials);
    }

    reduce_kernel<<<1, 256, 0, stream>>>(partials, scalar);
    add_geom_kernel<<<(N_ + 255) / 256, 256, 0, stream>>>(out, scalar);
}
